// MGGNN_78761110274184
// MI455X (gfx1250) — matrix operand order repaired and checked by reading, not compiled, not run on hardware
//
#include <hip/hip_runtime.h>

// ---------------------------------------------------------------------------
// MGGNN forward for gfx1250 (MI455X).  Wave32, WMMA f32_16x16x32_f16 GEMMs.
// Weights are pre-swizzled in LDS into per-lane-contiguous WMMA B-fragments
// (single v16h load -> 2x ds_load_b128); all B-fragments of a K-chunk are
// loaded before the WMMAs so the ds waits pipeline.
// ---------------------------------------------------------------------------

typedef __attribute__((ext_vector_type(16))) _Float16 v16h;
typedef __attribute__((ext_vector_type(8)))  _Float16 v8h;
typedef __attribute__((ext_vector_type(8)))  float    v8f;
typedef __attribute__((ext_vector_type(4)))  float    v4f;

#define BLK   256
#define NWAVE 8

// ----------------------------- device helpers ------------------------------

__device__ __forceinline__ v8f wmma32(v16h a, v16h b, v8f c) {
  // D = A(16x32 f16) * B(32x16 f16) + C(16x16 f32)
  return __builtin_amdgcn_wmma_f32_16x16x32_f16(false, a, false, b, (short)0, c,
                                                false, false);
}

__device__ __forceinline__ void atomAddF(float* p, float v) {
  __hip_atomic_fetch_add(p, v, __ATOMIC_RELAXED, __HIP_MEMORY_SCOPE_AGENT);
}

// Fill LDS with fp32 weights W[Kdim][Ndim] swizzled into B-fragment order:
// dst[((kc*(Ndim/16)+nc)*32 + lane)*16 + i] = W[kc*32 + (lane>>4)*8 + (i&7) + (i>>3)*16][nc*16 + (lane&15)]
// so a B-fragment is 16 contiguous halfs per lane (one aligned v16h load).
__device__ __forceinline__ void fill_bw(_Float16* dst, const float* W, int Kdim,
                                        int Ndim, int tid) {
  const int total = Kdim * Ndim;
  const int ncn = Ndim >> 4;
  for (int idx = tid; idx < total; idx += BLK) {
    int i    = idx & 15;
    int lane = (idx >> 4) & 31;
    int f    = idx >> 9;
    int nc   = f % ncn;
    int kc   = f / ncn;
    int K    = kc * 32 + (lane >> 4) * 8 + (i & 7) + (i >> 3) * 16;
    int col  = nc * 16 + (lane & 15);
    dst[idx] = (_Float16)W[(size_t)K * Ndim + col];
  }
}

__device__ __forceinline__ v16h b_frag(const _Float16* sW, int ncn, int kc,
                                       int nc, int lane) {
  return *(const v16h*)(sW + (((kc * ncn + nc) * 32 + lane) << 4));
}

// Issue all 4 B-fragment loads for a K-chunk, then the 4 WMMAs (pipelined waits)
__device__ __forceinline__ void wmma_n4(v16h a, const _Float16* sW, int kc,
                                        int lane, v8f* acc) {
  v16h b0 = b_frag(sW, 4, kc, 0, lane);
  v16h b1 = b_frag(sW, 4, kc, 1, lane);
  v16h b2 = b_frag(sW, 4, kc, 2, lane);
  v16h b3 = b_frag(sW, 4, kc, 3, lane);
  acc[0] = wmma32(a, b0, acc[0]);
  acc[1] = wmma32(a, b1, acc[1]);
  acc[2] = wmma32(a, b2, acc[2]);
  acc[3] = wmma32(a, b3, acc[3]);
}

__device__ __forceinline__ void wmma_n2(v16h a, const _Float16* sW, int kc,
                                        int lane, v8f* acc) {
  v16h b0 = b_frag(sW, 2, kc, 0, lane);
  v16h b1 = b_frag(sW, 2, kc, 1, lane);
  acc[0] = wmma32(a, b0, acc[0]);
  acc[1] = wmma32(a, b1, acc[1]);
}

// A-fragment (16x32 f16) from a global fp32 row pointer; two 8-elt runs.
// lane<16: h[0..7]=K(kb+0..7), h[8..15]=K(kb+16..23)
// lane>=16: h[0..7]=K(kb+8..15), h[8..15]=K(kb+24..31)
__device__ __forceinline__ v16h a_frag_global(const float* rowp, int kb, int hi) {
  v16h a;
  if (rowp) {
    const v4f* p = (const v4f*)(rowp + kb + hi * 8);
    v4f x0 = p[0], x1 = p[1], x2 = p[4], x3 = p[5];
#pragma unroll
    for (int i = 0; i < 4; i++) {
      a[i]      = (_Float16)x0[i];
      a[i + 4]  = (_Float16)x1[i];
      a[i + 8]  = (_Float16)x2[i];
      a[i + 12] = (_Float16)x3[i];
    }
  } else {
#pragma unroll
    for (int i = 0; i < 16; i++) a[i] = (_Float16)0.0f;
  }
  return a;
}

// A-fragment from LDS f16 row-major tile (ld halfs/row, 16B-aligned runs).
__device__ __forceinline__ v16h a_frag_lds(const _Float16* S, int ld, int row,
                                           int kb, int hi) {
  const _Float16* p = S + row * ld + kb + hi * 8;
  v8h lo = *(const v8h*)p;
  v8h hh = *(const v8h*)(p + 16);
  return __builtin_shufflevector(lo, hh, 0, 1, 2, 3, 4, 5, 6, 7, 8, 9, 10, 11,
                                 12, 13, 14, 15);
}

// ------------------------------ tiny kernels -------------------------------

__global__ void zero_kernel(float* __restrict__ p, size_t n) {
  size_t stride = (size_t)gridDim.x * blockDim.x;
  for (size_t g = (size_t)blockIdx.x * blockDim.x + threadIdx.x; g < n; g += stride)
    p[g] = 0.0f;
}

__global__ void relu_block_kernel(float* __restrict__ x, int N, int ld, int colOff) {
  size_t n = (size_t)N * 64;
  size_t stride = (size_t)gridDim.x * blockDim.x;
  for (size_t g = (size_t)blockIdx.x * blockDim.x + threadIdx.x; g < n; g += stride) {
    size_t row = g >> 6;
    int col = (int)(g & 63);
    float* p = x + row * ld + colOff + col;
    *p = fmaxf(*p, 0.0f);
  }
}

// hnext[dst[e]] += h[src[e]] * ew[e]   (segment_sum of messages)
__global__ void hop_kernel(const float* __restrict__ h, const int* __restrict__ src,
                           const int* __restrict__ dst, const float* __restrict__ ew,
                           float* __restrict__ hnext, int E, int D) {
  int D4 = D >> 2;
  size_t n = (size_t)E * D4;
  size_t stride = (size_t)gridDim.x * blockDim.x;
  for (size_t g = (size_t)blockIdx.x * blockDim.x + threadIdx.x; g < n; g += stride) {
    int e  = (int)(g / D4);
    int c0 = (int)(g % D4) << 2;
    float w = ew ? ew[e] : 1.0f;
    const v4f hv = *(const v4f*)(h + (size_t)src[e] * D + c0);
    float* op = hnext + (size_t)dst[e] * D + c0;
    atomAddF(op + 0, hv[0] * w);
    atomAddF(op + 1, hv[1] * w);
    atomAddF(op + 2, hv[2] * w);
    atomAddF(op + 3, hv[3] * w);
  }
}

__global__ void colstats_kernel(const float* __restrict__ x, int N,
                                float* __restrict__ stats /*[128]*/) {
  int col  = threadIdx.x & 63;
  int rgrp = threadIdx.x >> 6;  // 4 row groups per block
  float s = 0.0f, s2 = 0.0f;
  for (int r = blockIdx.x * 4 + rgrp; r < N; r += gridDim.x * 4) {
    float v = x[(size_t)r * 64 + col];
    s += v;
    s2 += v * v;
  }
  atomAddF(&stats[col], s);
  atomAddF(&stats[64 + col], s2);
}

__global__ void inorm_apply_kernel(float* __restrict__ x, int N,
                                   const float* __restrict__ stats) {
  size_t n = (size_t)N * 64;
  size_t stride = (size_t)gridDim.x * blockDim.x;
  float invN = 1.0f / (float)N;
  for (size_t g = (size_t)blockIdx.x * blockDim.x + threadIdx.x; g < n; g += stride) {
    int col = (int)(g & 63);
    float mu  = stats[col] * invN;
    float var = stats[64 + col] * invN - mu * mu;
    x[g] = (x[g] - mu) * rsqrtf(var + 1e-5f);
  }
}

__global__ void colsum_kernel(const float* __restrict__ vals,
                              const int* __restrict__ dst, int E,
                              float* __restrict__ cs) {
  size_t stride = (size_t)gridDim.x * blockDim.x;
  for (size_t g = (size_t)blockIdx.x * blockDim.x + threadIdx.x; g < (size_t)E; g += stride)
    atomAddF(&cs[dst[g]], vals[g]);
}

__global__ void rdiv_kernel(const float* __restrict__ vals,
                            const int* __restrict__ dst, int E,
                            const float* __restrict__ cs, float* __restrict__ out) {
  size_t stride = (size_t)gridDim.x * blockDim.x;
  for (size_t g = (size_t)blockIdx.x * blockDim.x + threadIdx.x; g < (size_t)E; g += stride)
    out[g] = vals[g] / cs[dst[g]];
}

// --------------------------- WMMA GEMM (M x K @ K x 64) --------------------
// flags: 1 = add bias, 2 = accumulate into existing out, 4 = relu on final sum
__global__ void __launch_bounds__(BLK)
gemm_kernel(const float* __restrict__ A, int M, int K,
            const float* __restrict__ W, const float* __restrict__ bias,
            float* __restrict__ out, int ldOut, int colOff, int flags) {
  __shared__ __align__(32) _Float16 sW[128 * 64];
  __shared__ float sB[64];
  const int tid = threadIdx.x;
  fill_bw(sW, W, K, 64, tid);
  if (tid < 64) sB[tid] = (flags & 1) ? bias[tid] : 0.0f;
  __syncthreads();

  const int lane = tid & 31, wave = tid >> 5;
  const int l15 = lane & 15, hi = lane >> 4;
  const int numTiles = (M + 15) >> 4;
  for (int t = blockIdx.x * NWAVE + wave; t < numTiles; t += gridDim.x * NWAVE) {
    const int m0 = t << 4;
    const int row = m0 + l15;
    const float* ap = (row < M) ? (A + (size_t)row * K) : nullptr;
    v8f z = {};
    v8f acc[4] = {z, z, z, z};
    for (int kc = 0; kc < (K >> 5); kc++) {
      v16h a = a_frag_global(ap, kc * 32, hi);
      wmma_n4(a, sW, kc, lane, acc);
    }
#pragma unroll
    for (int n = 0; n < 4; n++) {
      const int col = n * 16 + l15;
#pragma unroll
      for (int r = 0; r < 8; r++) {
        const int rr = m0 + hi * 8 + r;
        if (rr < M) {
          float v = acc[n][r] + sB[col];
          float* op = out + (size_t)rr * ldOut + colOff + col;
          if (flags & 2) v += *op;
          if (flags & 4) v = fmaxf(v, 0.0f);
          *op = v;
        }
      }
    }
  }
}

// ---------------- fused node MLP: [1 -> 64 relu -> 64 relu -> 64] ----------
__global__ void __launch_bounds__(BLK)
node_mlp_kernel(const float* __restrict__ x, int N,
                const float* __restrict__ W1, const float* __restrict__ b1,
                const float* __restrict__ W2, const float* __restrict__ b2,
                const float* __restrict__ W3, const float* __restrict__ b3,
                float* __restrict__ out) {
  __shared__ float sW1[64], sB1[64], sB2[64], sB3[64];
  __shared__ __align__(32) _Float16 sW2[64 * 64];
  __shared__ __align__(32) _Float16 sW3[64 * 64];
  __shared__ __align__(32) _Float16 sS[NWAVE][16 * 64];
  const int tid = threadIdx.x;
  if (tid < 64) { sW1[tid] = W1[tid]; sB1[tid] = b1[tid]; sB2[tid] = b2[tid]; sB3[tid] = b3[tid]; }
  fill_bw(sW2, W2, 64, 64, tid);
  fill_bw(sW3, W3, 64, 64, tid);
  __syncthreads();

  const int lane = tid & 31, wave = tid >> 5;
  const int l15 = lane & 15, hi = lane >> 4;
  _Float16* S = sS[wave];
  const int numTiles = (N + 15) >> 4;
  for (int t = blockIdx.x * NWAVE + wave; t < numTiles; t += gridDim.x * NWAVE) {
    const int m0 = t << 4;
    // layer 1 (scalar input) -> staging; contiguous 32-half run per lane
    {
      const int base = lane * 32;       // row = lane/2, cols (lane&1)*32..+31
      const int rr = m0 + (base >> 6);
      const int col0 = base & 63;
      float xv = (rr < N) ? x[rr] : 0.0f;
#pragma unroll
      for (int j = 0; j < 32; j++) {
        int col = col0 + j;
        S[base + j] = (_Float16)fmaxf(xv * sW1[col] + sB1[col], 0.0f);
      }
    }
    v8f z = {};
    // layer 2: WMMA, relu, back into staging
    {
      v8f acc[4] = {z, z, z, z};
#pragma unroll
      for (int kc = 0; kc < 2; kc++) {
        v16h a = a_frag_lds(S, 64, l15, kc * 32, hi);
        wmma_n4(a, sW2, kc, lane, acc);
      }
#pragma unroll
      for (int n = 0; n < 4; n++) {
        const int col = n * 16 + l15;
#pragma unroll
        for (int r = 0; r < 8; r++)
          S[(hi * 8 + r) * 64 + col] = (_Float16)fmaxf(acc[n][r] + sB2[col], 0.0f);
      }
    }
    // layer 3: WMMA, linear, write global
    {
      v8f acc[4] = {z, z, z, z};
#pragma unroll
      for (int kc = 0; kc < 2; kc++) {
        v16h a = a_frag_lds(S, 64, l15, kc * 32, hi);
        wmma_n4(a, sW3, kc, lane, acc);
      }
#pragma unroll
      for (int n = 0; n < 4; n++) {
        const int col = n * 16 + l15;
#pragma unroll
        for (int r = 0; r < 8; r++) {
          const int rr = m0 + hi * 8 + r;
          if (rr < N) out[(size_t)rr * 64 + col] = acc[n][r] + sB3[col];
        }
      }
    }
  }
}

// -------- fused scalar MLP: [KIN -> 64 relu -> 64 relu -> 1] ---------------
template <int KIN>
__global__ void __launch_bounds__(BLK)
scalar_mlp_kernel(const float* __restrict__ in, int E,
                  const float* __restrict__ W1, const float* __restrict__ b1,
                  const float* __restrict__ W2, const float* __restrict__ b2,
                  const float* __restrict__ W3, const float* __restrict__ b3,
                  float* __restrict__ out) {
  __shared__ float sW1[KIN * 64], sB1[64], sB2[64], sW3[64], sB3s;
  __shared__ __align__(32) _Float16 sW2[64 * 64];
  __shared__ __align__(32) _Float16 sS[NWAVE][16 * 64];
  const int tid = threadIdx.x;
  for (int i = tid; i < KIN * 64; i += BLK) sW1[i] = W1[i];
  if (tid < 64) { sB1[tid] = b1[tid]; sB2[tid] = b2[tid]; sW3[tid] = W3[tid]; }
  fill_bw(sW2, W2, 64, 64, tid);
  if (tid == 0) sB3s = b3[0];
  __syncthreads();

  const int lane = tid & 31, wave = tid >> 5;
  const int l15 = lane & 15, hi = lane >> 4;
  _Float16* S = sS[wave];
  const int numTiles = (E + 15) >> 4;
  for (int t = blockIdx.x * NWAVE + wave; t < numTiles; t += gridDim.x * NWAVE) {
    const int e0 = t << 4;
    // layer 1 -> staging; contiguous 32-half run per lane
    {
      const int base = lane * 32;
      const int rit = base >> 6;
      const int ee = e0 + rit;
      const int col0 = base & 63;
      float xin[KIN];
#pragma unroll
      for (int k = 0; k < KIN; k++)
        xin[k] = (ee < E) ? in[(size_t)ee * KIN + k] : 0.0f;
#pragma unroll
      for (int j = 0; j < 32; j++) {
        int col = col0 + j;
        float s = (ee < E) ? sB1[col] : 0.0f;
#pragma unroll
        for (int k = 0; k < KIN; k++) s += xin[k] * sW1[k * 64 + col];
        S[base + j] = (_Float16)fmaxf(s, 0.0f);
      }
    }
    v8f z = {};
    v8f acc[4] = {z, z, z, z};
#pragma unroll
    for (int kc = 0; kc < 2; kc++) {
      v16h a = a_frag_lds(S, 64, l15, kc * 32, hi);
      wmma_n4(a, sW2, kc, lane, acc);
    }
#pragma unroll
    for (int n = 0; n < 4; n++) {
      const int col = n * 16 + l15;
#pragma unroll
      for (int r = 0; r < 8; r++)
        S[(hi * 8 + r) * 64 + col] = (_Float16)fmaxf(acc[n][r] + sB2[col], 0.0f);
    }
    // final 64 -> 1 dot, lanes 0..15 handle the 16 rows (vector loads)
    if (hi == 0) {
      int ee = e0 + l15;
      if (ee < E) {
        float s = sB3s;
        const v8h* sp = (const v8h*)(S + l15 * 64);
#pragma unroll
        for (int c = 0; c < 8; c++) {
          v8h hv = sp[c];
#pragma unroll
          for (int k = 0; k < 8; k++) s += (float)hv[k] * sW3[c * 8 + k];
        }
        out[ee] = s;
      }
    }
  }
}

// --- fused edge message MLP: cat(Xa[idxA],Xb[idxB]) [128->64 relu ->64],
//     scatter-add at idxA into out[:, colOff:colOff+64] ----------------------
__global__ void __launch_bounds__(BLK)
edge_mp_kernel(const float* __restrict__ Xa, const int* __restrict__ idxA,
               const float* __restrict__ Xb, const int* __restrict__ idxB, int E,
               const float* __restrict__ W1, const float* __restrict__ b1,
               const float* __restrict__ W2, const float* __restrict__ b2,
               float* __restrict__ out, int ldOut, int colOff) {
  __shared__ __align__(32) _Float16 sW1[128 * 64];
  __shared__ __align__(32) _Float16 sW2[64 * 64];
  __shared__ float sB1[64], sB2[64];
  __shared__ __align__(32) _Float16 sS[NWAVE][16 * 64];
  const int tid = threadIdx.x;
  fill_bw(sW1, W1, 128, 64, tid);
  fill_bw(sW2, W2, 64, 64, tid);
  if (tid < 64) { sB1[tid] = b1[tid]; sB2[tid] = b2[tid]; }
  __syncthreads();

  const int lane = tid & 31, wave = tid >> 5;
  const int l15 = lane & 15, hi = lane >> 4;
  _Float16* S = sS[wave];
  const int numTiles = (E + 15) >> 4;
  for (int t = blockIdx.x * NWAVE + wave; t < numTiles; t += gridDim.x * NWAVE) {
    const int e0 = t << 4;
    const int e = e0 + l15;
    const bool valid = e < E;
    const float* fa = valid ? Xa + (size_t)idxA[e] * 64 : nullptr;
    const float* fb = valid ? Xb + (size_t)idxB[e] * 64 : nullptr;
    v8f z = {};
    v8f acc[4] = {z, z, z, z};
#pragma unroll
    for (int kc = 0; kc < 4; kc++) {
      const float* src = (kc < 2) ? (fa ? fa + kc * 32 : nullptr)
                                  : (fb ? fb + (kc - 2) * 32 : nullptr);
      v16h a = a_frag_global(src, 0, hi);
      wmma_n4(a, sW1, kc, lane, acc);
    }
#pragma unroll
    for (int n = 0; n < 4; n++) {
      const int col = n * 16 + l15;
#pragma unroll
      for (int r = 0; r < 8; r++)
        S[(hi * 8 + r) * 64 + col] = (_Float16)fmaxf(acc[n][r] + sB1[col], 0.0f);
    }
    v8f acc2[4] = {z, z, z, z};
#pragma unroll
    for (int kc = 0; kc < 2; kc++) {
      v16h a = a_frag_lds(S, 64, l15, kc * 32, hi);
      wmma_n4(a, sW2, kc, lane, acc2);
    }
#pragma unroll
    for (int n = 0; n < 4; n++) {
      const int col = n * 16 + l15;
#pragma unroll
      for (int r = 0; r < 8; r++) {
        const int ee = e0 + hi * 8 + r;
        if (ee < E) {
          const int d = idxA[ee];
          atomAddF(out + (size_t)d * ldOut + colOff + col, acc2[n][r] + sB2[col]);
        }
      }
    }
  }
}

// --- fused edge scalar MLP: cat(Xa[idxA],Xb[idxB])
//     [128->64 relu ->32 relu ->16 relu ->1], write out[e] -------------------
__global__ void __launch_bounds__(BLK)
edge_scalar_mlp_kernel(const float* __restrict__ Xa, const int* __restrict__ idxA,
                       const float* __restrict__ Xb, const int* __restrict__ idxB,
                       int E,
                       const float* __restrict__ W1, const float* __restrict__ b1,
                       const float* __restrict__ W2, const float* __restrict__ b2,
                       const float* __restrict__ W3, const float* __restrict__ b3,
                       const float* __restrict__ W4, const float* __restrict__ b4,
                       float* __restrict__ out) {
  __shared__ __align__(32) _Float16 sW1[128 * 64];
  __shared__ __align__(32) _Float16 sW2[64 * 32];
  __shared__ __align__(32) _Float16 sW3[32 * 16];
  __shared__ float sW4[16], sB1[64], sB2[32], sB3[16], sB4s;
  __shared__ __align__(32) _Float16 sS1[NWAVE][16 * 64];
  __shared__ __align__(32) _Float16 sS2[NWAVE][16 * 32];
  __shared__ __align__(16) float sS3[NWAVE][16 * 16];
  const int tid = threadIdx.x;
  fill_bw(sW1, W1, 128, 64, tid);
  fill_bw(sW2, W2, 64, 32, tid);
  fill_bw(sW3, W3, 32, 16, tid);
  if (tid < 64) sB1[tid] = b1[tid];
  if (tid < 32) sB2[tid] = b2[tid];
  if (tid < 16) { sB3[tid] = b3[tid]; sW4[tid] = W4[tid]; }
  if (tid == 0) sB4s = b4[0];
  __syncthreads();

  const int lane = tid & 31, wave = tid >> 5;
  const int l15 = lane & 15, hi = lane >> 4;
  _Float16* S1 = sS1[wave];
  _Float16* S2 = sS2[wave];
  float* S3 = sS3[wave];
  const int numTiles = (E + 15) >> 4;
  for (int t = blockIdx.x * NWAVE + wave; t < numTiles; t += gridDim.x * NWAVE) {
    const int e0 = t << 4;
    const int e = e0 + l15;
    const bool valid = e < E;
    const float* fa = valid ? Xa + (size_t)idxA[e] * 64 : nullptr;
    const float* fb = valid ? Xb + (size_t)idxB[e] * 64 : nullptr;
    v8f z = {};
    // layer 1: K=128, N=64
    v8f acc[4] = {z, z, z, z};
#pragma unroll
    for (int kc = 0; kc < 4; kc++) {
      const float* src = (kc < 2) ? (fa ? fa + kc * 32 : nullptr)
                                  : (fb ? fb + (kc - 2) * 32 : nullptr);
      v16h a = a_frag_global(src, 0, hi);
      wmma_n4(a, sW1, kc, lane, acc);
    }
#pragma unroll
    for (int n = 0; n < 4; n++) {
      const int col = n * 16 + l15;
#pragma unroll
      for (int r = 0; r < 8; r++)
        S1[(hi * 8 + r) * 64 + col] = (_Float16)fmaxf(acc[n][r] + sB1[col], 0.0f);
    }
    // layer 2: K=64, N=32
    v8f acc2[2] = {z, z};
#pragma unroll
    for (int kc = 0; kc < 2; kc++) {
      v16h a = a_frag_lds(S1, 64, l15, kc * 32, hi);
      wmma_n2(a, sW2, kc, lane, acc2);
    }
#pragma unroll
    for (int n = 0; n < 2; n++) {
      const int col = n * 16 + l15;
#pragma unroll
      for (int r = 0; r < 8; r++)
        S2[(hi * 8 + r) * 32 + col] = (_Float16)fmaxf(acc2[n][r] + sB2[col], 0.0f);
    }
    // layer 3: K=32, N=16
    v8f acc3 = z;
    {
      v16h a = a_frag_lds(S2, 32, l15, 0, hi);
      acc3 = wmma32(a, b_frag(sW3, 1, 0, 0, lane), acc3);
    }
    {
      const int col = l15;
#pragma unroll
      for (int r = 0; r < 8; r++)
        S3[(hi * 8 + r) * 16 + col] = fmaxf(acc3[r] + sB3[col], 0.0f);
    }
    // layer 4: 16 -> 1 dot (vector loads)
    if (hi == 0) {
      int ee = e0 + l15;
      if (ee < E) {
        float s = sB4s;
        const v4f* sp = (const v4f*)(S3 + l15 * 16);
#pragma unroll
        for (int c = 0; c < 4; c++) {
          v4f fv = sp[c];
#pragma unroll
          for (int k = 0; k < 4; k++) s += fv[k] * sW4[c * 4 + k];
        }
        out[ee] = s;
      }
    }
  }
}

// ------------------------------- host side ---------------------------------
// Param leaf order assumes JAX pytree flattening (dict keys sorted):
// edge_model(8), edge_model_R(8), layers[0..1](24 each:
//   mp01 W0,b0,W1,b1 | mp10 | tag_fc0 bias,l0,l1,l2 | tag_fc1 | tag_ff0 | tag_ff1),
// lin_out(2), lin_out_c(2), pre_edge(6), pre_edge_main(6), pre_node(6)

extern "C" void kernel_launch(void* const* d_in, const int* in_sizes, int n_in,
                              void* d_out, int out_size, void* d_ws, size_t ws_size,
                              hipStream_t stream) {
  if (n_in < 100) return;

  const float* x0     = (const float*)d_in[0];
  const float* x1     = (const float*)d_in[1];
  const float* attr00 = (const float*)d_in[2];
  const float* attr11 = (const float*)d_in[3];
  const int* e00s = (const int*)d_in[4];
  const int* e00d = (const int*)d_in[5];
  const int* e11s = (const int*)d_in[6];
  const int* e11d = (const int*)d_in[7];
  const int* e01s = (const int*)d_in[8];
  const int* e01d = (const int*)d_in[9];
  const int* e10s = (const int*)d_in[10];
  const int* e10d = (const int*)d_in[11];
  const int* msrc = (const int*)d_in[12];
  const int* mdst = (const int*)d_in[13];
  auto F = [&](int i) { return (const float*)d_in[i]; };

  const int N0 = in_sizes[0], N1 = in_sizes[1];
  const int E00 = in_sizes[4], E11 = in_sizes[6];
  const int E01 = in_sizes[8], E10 = in_sizes[10], EM = in_sizes[12];

  // workspace carve (fp32), 64-float (256B) aligned chunks
  float* ws = (float*)d_ws;
  size_t off = 0;
  auto walloc = [&](size_t n) {
    float* p = ws + off;
    off += (n + 63) & ~(size_t)63;
    return p;
  };
  float* h0    = walloc((size_t)N0 * 64);
  float* h1    = walloc((size_t)N1 * 64);
  float* a00t  = walloc((size_t)E00);
  float* a00   = walloc((size_t)E00);
  float* a11   = walloc((size_t)E11);
  float* c0    = walloc((size_t)N0 * 128);
  float* c1    = walloc((size_t)N1 * 128);
  float* hopA  = walloc((size_t)N0 * 128);
  float* hopB  = walloc((size_t)N0 * 128);
  float* bufX  = walloc((size_t)N0 * 64);
  float* bufXc = walloc((size_t)N1 * 64);
  float* Rraw  = walloc((size_t)E10);
  float* csum  = walloc((size_t)N0);
  float* stats = walloc(128);
  if (off * sizeof(float) > ws_size) return;

  auto tgrid = [](int M) {
    int t = (M + 15) / 16;
    int b = (t + NWAVE - 1) / NWAVE;
    if (b > 3072) b = 3072;
    if (b < 1) b = 1;
    return b;
  };
  auto fgrid = [](size_t n) {
    size_t b = (n + BLK - 1) / BLK;
    if (b > 4096) b = 4096;
    if (b < 1) b = 1;
    return (int)b;
  };
  auto zero = [&](float* p, size_t n) { zero_kernel<<<fgrid(n), BLK, 0, stream>>>(p, n); };
  auto gemm = [&](const float* A, int M, int K, const float* W, const float* b,
                  float* o, int ld, int co, int fl) {
    gemm_kernel<<<tgrid(M), BLK, 0, stream>>>(A, M, K, W, b, o, ld, co, fl);
  };
  // TAGConv (K=2): out[:,co:co+64] = relu(x@l0 + A(x)@l1 + A(A(x))@l2 + bias)
  auto tag = [&](const float* xin, int Nn, int Din, int E, const int* src,
                 const int* dst, const float* ew, const float* bias,
                 const float* l0, const float* l1, const float* l2, float* outp,
                 int ld, int co) {
    gemm(xin, Nn, Din, l0, bias, outp, ld, co, 1);
    zero(hopA, (size_t)Nn * Din);
    hop_kernel<<<fgrid((size_t)E * (Din / 4)), BLK, 0, stream>>>(xin, src, dst, ew,
                                                                 hopA, E, Din);
    gemm(hopA, Nn, Din, l1, nullptr, outp, ld, co, 2);
    zero(hopB, (size_t)Nn * Din);
    hop_kernel<<<fgrid((size_t)E * (Din / 4)), BLK, 0, stream>>>(hopA, src, dst, ew,
                                                                 hopB, E, Din);
    gemm(hopB, Nn, Din, l2, nullptr, outp, ld, co, 2 | 4);  // accum + final relu
  };
  auto inorm = [&](float* xp, int Nn) {
    zero(stats, 128);
    colstats_kernel<<<512, BLK, 0, stream>>>(xp, Nn, stats);
    inorm_apply_kernel<<<fgrid((size_t)Nn * 64), BLK, 0, stream>>>(xp, Nn, stats);
  };

  // ---- pre stage ----
  node_mlp_kernel<<<tgrid(N0), BLK, 0, stream>>>(x0, N0, F(94), F(95), F(96), F(97),
                                                 F(98), F(99), h0);
  node_mlp_kernel<<<tgrid(N1), BLK, 0, stream>>>(x1, N1, F(94), F(95), F(96), F(97),
                                                 F(98), F(99), h1);
  scalar_mlp_kernel<2><<<tgrid(E00), BLK, 0, stream>>>(attr00, E00, F(88), F(89),
                                                       F(90), F(91), F(92), F(93), a00t);
  scalar_mlp_kernel<1><<<tgrid(E00), BLK, 0, stream>>>(a00t, E00, F(82), F(83),
                                                       F(84), F(85), F(86), F(87), a00);
  scalar_mlp_kernel<1><<<tgrid(E11), BLK, 0, stream>>>(attr11, E11, F(82), F(83),
                                                       F(84), F(85), F(86), F(87), a11);

  // ---- message-passing layers ----
  for (int L = 0; L < 2; L++) {
    const int B = 30 + 24 * L;
    zero(c0, (size_t)N0 * 128);
    zero(c1, (size_t)N1 * 128);
    // o0_L0 = relu(tag_fc0(h0)) -> c0[:, 0:64]
    tag(h0, N0, 64, E00, e00s, e00d, nullptr, F(B + 8), F(B + 9), F(B + 10),
        F(B + 11), c0, 128, 0);
    // o1_L0 = relu(mp10(cat(h0[dst],h1[src]))) -> c0[:, 64:128]
    edge_mp_kernel<<<tgrid(E10), BLK, 0, stream>>>(h0, e10d, h1, e10s, E10,
                                                   F(B + 4), F(B + 5), F(B + 6),
                                                   F(B + 7), c0, 128, 64);
    relu_block_kernel<<<fgrid((size_t)N0 * 64), BLK, 0, stream>>>(c0, N0, 128, 64);
    // o0_L1 = relu(mp01(cat(h1[dst],h0[src]))) -> c1[:, 0:64]
    edge_mp_kernel<<<tgrid(E01), BLK, 0, stream>>>(h1, e01d, h0, e01s, E01,
                                                   F(B + 0), F(B + 1), F(B + 2),
                                                   F(B + 3), c1, 128, 0);
    relu_block_kernel<<<fgrid((size_t)N1 * 64), BLK, 0, stream>>>(c1, N1, 128, 0);
    // o1_L1 = relu(tag_fc1(h1)) -> c1[:, 64:128]
    tag(h1, N1, 64, E11, e11s, e11d, nullptr, F(B + 12), F(B + 13), F(B + 14),
        F(B + 15), c1, 128, 64);
    // h0 = inorm(relu(tag_ff0(c0, ew=a00)))
    tag(c0, N0, 128, E00, e00s, e00d, a00, F(B + 16), F(B + 17), F(B + 18),
        F(B + 19), h0, 64, 0);
    inorm(h0, N0);
    // h1 = inorm(relu(tag_ff1(c1, ew=a11)))
    tag(c1, N1, 128, E11, e11s, e11d, a11, F(B + 20), F(B + 21), F(B + 22),
        F(B + 23), h1, 64, 0);
    inorm(h1, N1);
  }

  // ---- output heads ----
  gemm(h0, N0, 64, F(78), F(79), bufX, 64, 0, 1);
  gemm(h1, N1, 64, F(80), F(81), bufXc, 64, 0, 1);

  float* outA = (float*)d_out;        // A_vals: EMASK
  float* outR = outA + EM;            // R_vals: E10
  // A_vals = edge_model(cat(x[mask_src], x[mask_dst]))
  edge_scalar_mlp_kernel<<<tgrid(EM), BLK, 0, stream>>>(
      bufX, msrc, bufX, mdst, EM, F(14), F(15), F(16), F(17), F(18), F(19), F(20),
      F(21), outA);
  // R_raw = edge_model_R(cat(xc[e10_src], x[e10_dst])); column-normalize
  edge_scalar_mlp_kernel<<<tgrid(E10), BLK, 0, stream>>>(
      bufXc, e10s, bufX, e10d, E10, F(22), F(23), F(24), F(25), F(26), F(27), F(28),
      F(29), Rraw);
  zero(csum, (size_t)N0);
  colsum_kernel<<<fgrid((size_t)E10), BLK, 0, stream>>>(Rraw, e10d, E10, csum);
  rdiv_kernel<<<fgrid((size_t)E10), BLK, 0, stream>>>(Rraw, e10d, E10, csum, outR);
}
